// Dtu_32280974197058
// MI455X (gfx1250) — matrix operand order repaired and checked by reading, not compiled, not run on hardware
//
#include <hip/hip_runtime.h>
#include <math.h>

#define EMBED  192
#define D1     576
#define NSP    56
#define BSZ    8
#define MROWS  (BSZ * NSP * NSP)   // 25088
#define TWO_N  (2 * NSP)           // 112
#define RPEH   32
#define NLAYER 3
#define LAMBDA 0.95f
#define CHB    64                  // channels per Toeplitz block

typedef __attribute__((ext_vector_type(16))) __bf16 v16bf;
typedef __attribute__((ext_vector_type(8)))  float  v8f;

union FragAB { v16bf v; __bf16 s[16]; };
union FragC  { v8f   v; float  s[8];  };

// Fast silu: x * v_rcp_f32(1 + exp(-x)) — avoids the IEEE div_scale/refine chain.
__device__ __forceinline__ float silu_f(float x) {
    return x * __builtin_amdgcn_rcpf(1.0f + __expf(-x));
}

// ---- WMMA fragment loaders (ISA 7.12.2 bf16 layouts, wave32) -------------
// A 16x32 (MxK): lanes 0-15 -> M=lane, K = {0..7, 16..23}; lanes 16-31 ->
// M=lane-16, K = {8..15, 24..31}. Element e: k = (e&8?16:0) + kb + (e&7).
__device__ __forceinline__ v16bf load_a_f32(const float* A, int ldk, int row,
                                            int k0, int lane) {
    FragAB f;
    const float* ap = A + (size_t)(row + (lane & 15)) * ldk + k0;
    const int kb = (lane >> 4) * 8;
#pragma unroll
    for (int e = 0; e < 16; ++e) {
        int k = ((e & 8) ? 16 : 0) + kb + (e & 7);
        f.s[e] = (__bf16)ap[k];
    }
    return f.v;
}

__device__ __forceinline__ v16bf load_a_bf16(const __bf16* A, int ldk, int row,
                                             int k0, int lane) {
    FragAB f;
    const __bf16* ap = A + (size_t)(row + (lane & 15)) * ldk + k0;
    const int kb = (lane >> 4) * 8;
#pragma unroll
    for (int e = 0; e < 16; ++e) {
        int k = ((e & 8) ? 16 : 0) + kb + (e & 7);
        f.s[e] = ap[k];
    }
    return f.v;
}

// B 32x16 (KxN): lanes 0-15 -> N=lane, K = {0..7, 16..23}; lanes 16-31 ->
// N=lane-16, K = {8..15, 24..31}. Weights stored (N,K) row-major, so
// B[k][n] = W[n*K+k]; element e: k = (e&8?16:0) + kb + (e&7).
__device__ __forceinline__ v16bf load_b_w(const float* W, int K, int col,
                                          int k0, int lane) {
    FragAB f;
    const float* wp = W + (size_t)(col + (lane & 15)) * K + k0 + ((lane >> 4) * 8);
#pragma unroll
    for (int e = 0; e < 16; ++e) f.s[e] = (__bf16)wp[((e & 8) ? 16 : 0) + (e & 7)];
    return f.v;
}

// ---- Generic WMMA GEMM: C = act(A @ W.T + bias) --------------------------
// Block: 128 threads = 4 waves. Each wave owns a 32x16 slab (two M-tiles
// sharing one B fragment per k-step -> 2 wmma per iteration, half the weight
// traffic). grid.x = M/32, grid.y = N/64. ACT: 0=none 1=silu. NT: streaming
// store (write-once output, keep L2 for the reused intermediates).
template <int ACT, bool ABF16, bool NT>
__global__ void wmma_gemm_kernel(const void* __restrict__ Aptr,
                                 const float* __restrict__ W,
                                 const float* __restrict__ bias,
                                 float* __restrict__ C, int N, int K) {
    const int lane    = threadIdx.x & 31;
    const int wave    = threadIdx.x >> 5;
    const int rowBase = blockIdx.x * 32;
    const int colBase = (blockIdx.y * 4 + wave) * 16;

    v8f acc0 = {};
    v8f acc1 = {};
    for (int k0 = 0; k0 < K; k0 += 32) {
        if (k0 + 32 < K)  // stream next weight chunk (global_prefetch_b8)
            __builtin_prefetch(W + (size_t)(colBase + (lane & 15)) * K + k0 + 32, 0, 0);
        v16bf b = load_b_w(W, K, colBase, k0, lane);
        v16bf a0, a1;
        if constexpr (ABF16) {
            a0 = load_a_bf16((const __bf16*)Aptr, K, rowBase,      k0, lane);
            a1 = load_a_bf16((const __bf16*)Aptr, K, rowBase + 16, k0, lane);
        } else {
            a0 = load_a_f32((const float*)Aptr, K, rowBase,      k0, lane);
            a1 = load_a_f32((const float*)Aptr, K, rowBase + 16, k0, lane);
        }
        acc0 = __builtin_amdgcn_wmma_f32_16x16x32_bf16(false, a0, false, b,
                                                       (short)0, acc0, false, false);
        acc1 = __builtin_amdgcn_wmma_f32_16x16x32_bf16(false, a1, false, b,
                                                       (short)0, acc1, false, false);
    }

    const int col = colBase + (lane & 15);
    const int rh  = (lane >> 4) * 8;          // C layout: lanes 16-31 hold M=8..15
    const float bv = bias[col];
#pragma unroll
    for (int t = 0; t < 2; ++t) {
        FragC c; c.v = t ? acc1 : acc0;
        const int rbase = rowBase + t * 16 + rh;
#pragma unroll
        for (int r = 0; r < 8; ++r) {
            float v = c.s[r] + bv;
            if constexpr (ACT == 1) v = silu_f(v);
            float* dst = &C[(size_t)(rbase + r) * N + col];
            if constexpr (NT) __builtin_nontemporal_store(v, dst);
            else              *dst = v;
        }
    }
}

// ---- Fused QV GEMM: QV = silu(X@Wq.T+bq) * silu(X@Wv.T+bv) ---------------
// Block: 64 threads = 2 waves; wave0 -> Q tile, wave1 -> V tile (same 16x16
// region), product exchanged through a 1 KB LDS tile. q never touches HBM.
__global__ void qv_gemm_kernel(const float* __restrict__ X,
                               const float* __restrict__ Wq, const float* __restrict__ bq,
                               const float* __restrict__ Wv, const float* __restrict__ bv,
                               float* __restrict__ QV) {
    __shared__ float tile[16 * 16];
    const int lane    = threadIdx.x & 31;
    const int wave    = threadIdx.x >> 5;
    const int rowBase = blockIdx.x * 16;
    const int colBase = blockIdx.y * 16;
    const float* Wm = wave ? Wv : Wq;
    const float* bm = wave ? bv : bq;

    v8f acc = {};
    for (int k0 = 0; k0 < EMBED; k0 += 32) {
        v16bf a = load_a_f32(X, EMBED, rowBase, k0, lane);
        v16bf b = load_b_w(Wm, EMBED, colBase, k0, lane);
        acc = __builtin_amdgcn_wmma_f32_16x16x32_bf16(false, a, false, b,
                                                      (short)0, acc, false, false);
    }

    FragC c; c.v = acc;
    const int col = lane & 15;
    const int rh  = (lane >> 4) * 8;
    const float bb = bm[colBase + col];
    float vals[8];
#pragma unroll
    for (int r = 0; r < 8; ++r) vals[r] = silu_f(c.s[r] + bb);

    if (wave == 1) {
#pragma unroll
        for (int r = 0; r < 8; ++r) tile[(rh + r) * 16 + col] = vals[r];
    }
    __syncthreads();
    if (wave == 0) {
#pragma unroll
        for (int r = 0; r < 8; ++r)
            QV[(size_t)(rowBase + rh + r) * D1 + colBase + col] =
                vals[r] * tile[(rh + r) * 16 + col];
    }
}

// ---- RPE MLP + decay: A{1,2}[row, c] (112 x 576) -------------------------
// grid (112, 2): blockIdx.y selects parameter set. 64 threads/block.
__global__ void rpe_kernel(const float* w0a, const float* b0a, const float* wsa,
                           const float* bsa, const float* woa, const float* boa,
                           const float* w0b, const float* b0b, const float* wsb,
                           const float* bsb, const float* wob, const float* bob,
                           const float* __restrict__ slope,
                           float* __restrict__ A1, float* __restrict__ A2) {
    const int  row = blockIdx.x;
    const bool sec = (blockIdx.y == 1);
    const float* w0 = sec ? w0b : w0a;  const float* b0 = sec ? b0b : b0a;
    const float* ws = sec ? wsb : wsa;  const float* bs = sec ? bsb : bsa;
    const float* wo = sec ? wob : woa;  const float* bo = sec ? bob : boa;
    float* Out = sec ? A2 : A1;

    // rpe_input rows: [0, 1..55, 0, -1..-55]
    float t;
    if (row == 0 || row == NSP)      t = 0.0f;
    else if (row < NSP)              t = (float)row;
    else                             t = -(float)(row - NSP);
    // decay exponent rows: [0, 1..55, 0, 55..1]
    int ep;
    if (row == 0 || row == NSP)      ep = 0;
    else if (row < NSP)              ep = row;
    else                             ep = TWO_N - row;

    __shared__ float h[RPEH];
    const int tid = threadIdx.x;
    if (tid < RPEH) h[tid] = t * w0[tid] + b0[tid];
    __syncthreads();

    for (int L = 0; L < NLAYER; ++L) {
        float v = 0.0f;
        if (tid < RPEH) {
            const float* Wl = ws + (size_t)L * RPEH * RPEH + (size_t)tid * RPEH;
            for (int k = 0; k < RPEH; ++k) v = fmaf(fmaxf(h[k], 0.0f), Wl[k], v);
            v += bs[L * RPEH + tid];
        }
        __syncthreads();
        if (tid < RPEH) h[tid] = v;
        __syncthreads();
    }

    for (int c = tid; c < D1; c += blockDim.x) {
        const float* wr = wo + (size_t)c * RPEH;
        float acc = bo[c];
        for (int k = 0; k < RPEH; ++k) acc = fmaf(fmaxf(h[k], 0.0f), wr[k], acc);
        float sl = LAMBDA + (1.0f - LAMBDA) * fminf(fmaxf(slope[c], 0.0f), 1.0f);
        // sl in [0.95, 1]: sl^ep = exp(ep * log(sl)) — fast path, no powf.
        float d  = (ep == 0) ? 1.0f : __expf((float)ep * __logf(sl));
        Out[(size_t)row * D1 + c] = acc * d;
    }
}

// ---- Toeplitz along width: MIX = T1 * QV (per channel) -------------------
// Block: one (b,h) row x 64 channels; 56-tap FMA per output from LDS tiles.
__global__ void toeplitz_w_kernel(const float* __restrict__ QV,
                                  const float* __restrict__ A1,
                                  float* __restrict__ MIX) {
    __shared__ float qs[NSP * CHB];
    __shared__ float as[TWO_N * CHB];
    const int bh  = blockIdx.x;
    const int c0  = blockIdx.y * CHB;
    const int tid = threadIdx.x;

    for (int idx = tid; idx < NSP * CHB; idx += blockDim.x) {
        int j = idx >> 6, cc = idx & (CHB - 1);
        qs[idx] = QV[((size_t)bh * NSP + j) * D1 + c0 + cc];
    }
    for (int idx = tid; idx < TWO_N * CHB; idx += blockDim.x) {
        int l = idx >> 6, cc = idx & (CHB - 1);
        as[idx] = A1[(size_t)l * D1 + c0 + cc];
    }
    __syncthreads();

    for (int idx = tid; idx < NSP * CHB; idx += blockDim.x) {
        int i = idx >> 6, cc = idx & (CHB - 1);
        float acc = 0.0f;
        for (int j = 0; j < NSP; ++j) {
            int lag = i - j; if (lag < 0) lag += TWO_N;   // (i-j) mod 2n
            acc = fmaf(as[lag * CHB + cc], qs[j * CHB + cc], acc);
        }
        MIX[((size_t)bh * NSP + i) * D1 + c0 + cc] = acc;
    }
}

// ---- Toeplitz along height + fused epilogue: S = bf16(P * (o1 + o2)) -----
__global__ void toeplitz_h_kernel(const float* __restrict__ QV,
                                  const float* __restrict__ A2,
                                  const float* __restrict__ MIX,
                                  const float* __restrict__ P,
                                  __bf16* __restrict__ S) {
    __shared__ float qs[NSP * CHB];
    __shared__ float as[TWO_N * CHB];
    const int b   = blockIdx.x / NSP;
    const int w   = blockIdx.x % NSP;
    const int c0  = blockIdx.y * CHB;
    const int tid = threadIdx.x;

    for (int idx = tid; idx < NSP * CHB; idx += blockDim.x) {
        int j = idx >> 6, cc = idx & (CHB - 1);
        qs[idx] = QV[((size_t)(b * NSP + j) * NSP + w) * D1 + c0 + cc];
    }
    for (int idx = tid; idx < TWO_N * CHB; idx += blockDim.x) {
        int l = idx >> 6, cc = idx & (CHB - 1);
        as[idx] = A2[(size_t)l * D1 + c0 + cc];
    }
    __syncthreads();

    for (int idx = tid; idx < NSP * CHB; idx += blockDim.x) {
        int i = idx >> 6, cc = idx & (CHB - 1);
        float acc = 0.0f;
        for (int j = 0; j < NSP; ++j) {
            int lag = i - j; if (lag < 0) lag += TWO_N;
            acc = fmaf(as[lag * CHB + cc], qs[j * CHB + cc], acc);
        }
        size_t g = ((size_t)(b * NSP + i) * NSP + w) * D1 + c0 + cc;
        S[g] = (__bf16)(P[g] * (acc + MIX[g]));
    }
}

extern "C" void kernel_launch(void* const* d_in, const int* in_sizes, int n_in,
                              void* d_out, int out_size, void* d_ws, size_t ws_size,
                              hipStream_t stream) {
    const float* x     = (const float*)d_in[0];
    const float* p_w   = (const float*)d_in[1];
    const float* p_b   = (const float*)d_in[2];
    const float* q_w   = (const float*)d_in[3];
    const float* q_b   = (const float*)d_in[4];
    const float* v_w   = (const float*)d_in[5];
    const float* v_b   = (const float*)d_in[6];
    const float* o_w   = (const float*)d_in[7];
    const float* o_b   = (const float*)d_in[8];
    const float* slope = (const float*)d_in[9];
    const float* t1_w0 = (const float*)d_in[10];
    const float* t1_b0 = (const float*)d_in[11];
    const float* t1_ws = (const float*)d_in[12];
    const float* t1_bs = (const float*)d_in[13];
    const float* t1_wo = (const float*)d_in[14];
    const float* t1_bo = (const float*)d_in[15];
    const float* t2_w0 = (const float*)d_in[16];
    const float* t2_b0 = (const float*)d_in[17];
    const float* t2_ws = (const float*)d_in[18];
    const float* t2_bs = (const float*)d_in[19];
    const float* t2_wo = (const float*)d_in[20];
    const float* t2_bo = (const float*)d_in[21];

    // Workspace layout (~203 MB): A1|A2 | P | QV | MIX | S(bf16)
    char*   ws  = (char*)d_ws;
    float*  A1  = (float*)ws;                         // 112*576 f32
    float*  A2  = A1 + (size_t)TWO_N * D1;
    float*  P   = (float*)(ws + (1u << 19));          // 512 KB offset
    float*  QV  = P  + (size_t)MROWS * D1;
    float*  MIX = QV + (size_t)MROWS * D1;
    __bf16* S   = (__bf16*)(MIX + (size_t)MROWS * D1);

    // 1) Toeplitz coefficient tables (RPE MLP * decay), both param sets.
    rpe_kernel<<<dim3(TWO_N, 2), 64, 0, stream>>>(
        t1_w0, t1_b0, t1_ws, t1_bs, t1_wo, t1_bo,
        t2_w0, t2_b0, t2_ws, t2_bs, t2_wo, t2_bo, slope, A1, A2);

    // 2) P = silu(X @ p_w.T + p_b)  — WMMA bf16, f32 accum, reused -> RT policy.
    wmma_gemm_kernel<1, false, false><<<dim3(MROWS / 32, D1 / 64), 128, 0, stream>>>(
        x, p_w, p_b, P, D1, EMBED);

    // 3) QV = silu(q) * silu(v) fused — q never hits HBM.
    qv_gemm_kernel<<<dim3(MROWS / 16, D1 / 16), 64, 0, stream>>>(
        x, q_w, q_b, v_w, v_b, QV);

    // 4) o1: per-channel Toeplitz mix along width -> MIX.
    toeplitz_w_kernel<<<dim3(BSZ * NSP, D1 / CHB), 256, 0, stream>>>(QV, A1, MIX);

    // 5) o2 along height + fused S = bf16(P * (o1 + o2)).
    toeplitz_h_kernel<<<dim3(BSZ * NSP, D1 / CHB), 256, 0, stream>>>(QV, A2, MIX, P, S);

    // 6) out = S @ o_w.T + o_b — bf16 A straight from S; write-once output
    //    streamed past L2 with nontemporal stores.
    wmma_gemm_kernel<0, true, true><<<dim3(MROWS / 32, EMBED / 64), 128, 0, stream>>>(
        S, o_w, o_b, (float*)d_out, EMBED, D1);
}
